// GraphLayer_70463233458716
// MI455X (gfx1250) — matrix operand order repaired and checked by reading, not compiled, not run on hardware
//
#include <hip/hip_runtime.h>

#define DEV __device__ __forceinline__

typedef __attribute__((ext_vector_type(16))) _Float16 v16h;
typedef __attribute__((ext_vector_type(8)))  _Float16 v8h;
typedef __attribute__((ext_vector_type(8)))  float    v8f;
typedef __attribute__((ext_vector_type(4)))  float    v4f;

#define BB   8
#define NN   2048
#define DIN_ 300
#define DD   96
#define KP   320
#define ROWS (BB*NN)          // 16384 flattened (b,n) rows
#define WAVES 4
#define BLOCKT (WAVES*32)
#define GGRID (ROWS/(16*WAVES))  // 256

// ---------------- WMMA wrapper ----------------
DEV v8f wmma_f16(v16h a, v16h b, v8f c) {
  // D = A(16x32 f16) * B(32x16 f16) + C(16x16 f32)
  return __builtin_amdgcn_wmma_f32_16x16x32_f16(false, a, false, b, (short)0, c, false, false);
}

// ---------------- fragment helpers ----------------
// A-frag (16x32) from fp32 row-major src at (row0,k0); ld in elements.
// Halves 0..7 <-> K = 8*hi+0..7 ; halves 8..15 <-> K = 16+8*hi+0..7
DEV v16h load_afrag_f32(const float* base, int ld, int lo, int hi) {
  const float* p = base + (size_t)lo * ld;
  const int r0 = 8*hi, r1 = 16 + 8*hi;
  v4f x0 = *(const v4f*)(p + r0);
  v4f x1 = *(const v4f*)(p + r0 + 4);
  v4f x2 = *(const v4f*)(p + r1);
  v4f x3 = *(const v4f*)(p + r1 + 4);
  v16h a;
#pragma unroll
  for (int i=0;i<4;i++) { a[i]=(_Float16)x0[i]; a[4+i]=(_Float16)x1[i];
                          a[8+i]=(_Float16)x2[i]; a[12+i]=(_Float16)x3[i]; }
  return a;
}

DEV v16h load_afrag_f32_guard(const float* base, int ld, int lo, int hi, int k0, int kmax) {
  const float* p = base + (size_t)lo * ld;
  const int r0 = 8*hi, r1 = 16 + 8*hi;
  v16h a;
#pragma unroll
  for (int i=0;i<8;i++) {
    float v0 = 0.f, v1 = 0.f;
    if (k0 + r0 + i < kmax) v0 = p[r0 + i];
    if (k0 + r1 + i < kmax) v1 = p[r1 + i];
    a[i] = (_Float16)v0; a[8+i] = (_Float16)v1;
  }
  return a;
}

DEV v16h load_afrag_f16(const _Float16* base, int ld, int lo, int hi) {
  const _Float16* p = base + (size_t)lo * ld;
  v8h l0 = *(const v8h*)(p + 8*hi);
  v8h l1 = *(const v8h*)(p + 16 + 8*hi);
  v16h a;
#pragma unroll
  for (int i=0;i<8;i++){ a[i]=l0[i]; a[8+i]=l1[i]; }
  return a;
}

// B-frag (32x16) from f16 "BT" row-major [n][k] at (n0,k0); lane lo = column n.
// Halves 0..7 <-> K = 8*hi+0..7 ; halves 8..15 <-> K = 16+8*hi+0..7
DEV v16h load_bfrag_f16(const _Float16* base, int ld, int lo, int hi) {
  const _Float16* p = base + (size_t)lo * ld;
  v8h l0 = *(const v8h*)(p + 8*hi);
  v8h l1 = *(const v8h*)(p + 16 + 8*hi);
  v16h b;
#pragma unroll
  for (int i=0;i<8;i++){ b[i]=l0[i]; b[8+i]=l1[i]; }
  return b;
}

// C-frag (16x16 f32): VGPR j <-> row 8*hi+j, col = lane lo
DEV void store_cfrag_f32(float* dst, int ld, v8f c, int lo, int hi) {
  float* p = dst + (size_t)(8*hi)*ld + lo;
#pragma unroll
  for (int j=0;j<8;j++) p[(size_t)j*ld] = c[j];
}
DEV v8f load_cfrag_f32(const float* src, int ld, int lo, int hi) {
  const float* p = src + (size_t)(8*hi)*ld + lo;
  v8f c;
#pragma unroll
  for (int j=0;j<8;j++) c[j] = p[(size_t)j*ld];
  return c;
}

DEV float sigmoidf_(float v){ return 1.f/(1.f + __expf(-v)); }

// ---------------- weight prep: transpose + fp32->f16 ----------------
__global__ __launch_bounds__(256) void k_prep(
    const float* __restrict__ W_enc, const float* __restrict__ W_ii,
    const float* __restrict__ W_gru, const float* __restrict__ W_att,
    _Float16* __restrict__ WencT, _Float16* __restrict__ WiiT,
    _Float16* __restrict__ WgruT, _Float16* __restrict__ WattT)
{
  int idx = blockIdx.x*256 + threadIdx.x;
  const int nEnc = 3*DD*KP;    // padded K: 300 -> 320
  const int nGru = 18*DD*DD;
  const int nW   = 3*DD*DD;
  if (idx < nEnc) {
    int i = idx/(DD*KP); int r = idx%(DD*KP); int j = r/KP; int k = r%KP;
    WencT[idx] = (k < DIN_) ? (_Float16)W_enc[((size_t)i*DIN_ + k)*DD + j] : (_Float16)0.f;
  } else if ((idx -= nEnc) < nGru) {
    int ig = idx/(DD*DD); int r = idx%(DD*DD); int j = r/DD; int k = r%DD;
    WgruT[idx] = (_Float16)W_gru[((size_t)ig*DD + k)*DD + j];
  } else if ((idx -= nGru) < nW) {
    int i = idx/(DD*DD); int r = idx%(DD*DD); int j = r/DD; int k = r%DD;
    WiiT[idx] = (_Float16)W_ii[((size_t)i*DD + k)*DD + j];
  } else if ((idx -= nW) < nW) {
    int i = idx/(DD*DD); int r = idx%(DD*DD); int j = r/DD; int k = r%DD;
    WattT[idx] = (_Float16)W_att[((size_t)i*DD + k)*DD + j];
  }
}

// ---------------- encode: h = mask * relu(x @ Wenc + b) ----------------
__global__ __launch_bounds__(BLOCKT) void k_encode(
    const float* __restrict__ x, const float* __restrict__ mask,
    const _Float16* __restrict__ WencT, const float* __restrict__ benc,
    float* __restrict__ h, _Float16* __restrict__ hT)
{
  const int wave = threadIdx.x >> 5, lane = threadIdx.x & 31;
  const int lo = lane & 15, hi = lane >> 4;
  const int row0 = (blockIdx.x*WAVES + wave)*16;
  v8f acc[6];
#pragma unroll
  for (int f=0; f<6; f++) { float bv = benc[f*16+lo];
#pragma unroll
    for (int j=0;j<8;j++) acc[f][j] = bv; }
  const float* Ab = x + (size_t)row0 * DIN_;
#pragma unroll 1
  for (int kc=0; kc<9; kc++) {
    v16h a = load_afrag_f32(Ab + kc*32, DIN_, lo, hi);
#pragma unroll
    for (int f=0; f<6; f++)
      acc[f] = wmma_f16(a, load_bfrag_f16(WencT + (size_t)(f*16)*KP + kc*32, KP, lo, hi), acc[f]);
  }
  { // K remainder 288..299 (guarded loads; WencT padded with zeros)
    v16h a = load_afrag_f32_guard(Ab + 288, DIN_, lo, hi, 288, DIN_);
#pragma unroll
    for (int f=0; f<6; f++)
      acc[f] = wmma_f16(a, load_bfrag_f16(WencT + (size_t)(f*16)*KP + 288, KP, lo, hi), acc[f]);
  }
#pragma unroll
  for (int j=0;j<8;j++) {
    int m = row0 + 8*hi + j;
    float mv = mask[m];
    int bb = m >> 11, nn = m & (NN-1);
#pragma unroll
    for (int f=0; f<6; f++) {
      float v = acc[f][j]; v = v > 0.f ? v : 0.f; v *= mv;
      int d = f*16 + lo;
      h[(size_t)m*DD + d] = v;
      hT[((size_t)bb*DD + d)*NN + nn] = (_Float16)v;
    }
  }
}

// ---------------- message passing: a = adj @ h (adj fp32->f16 on the fly) ----------------
__global__ __launch_bounds__(BLOCKT) void k_mp(
    const float* __restrict__ adj, const _Float16* __restrict__ hT,
    float* __restrict__ aOut)
{
  const int wave = threadIdx.x >> 5, lane = threadIdx.x & 31;
  const int lo = lane & 15, hi = lane >> 4;
  const int row0 = (blockIdx.x*WAVES + wave)*16;
  const int bb = row0 >> 11, n0 = row0 & (NN-1);
  const float*    Ab = adj + ((size_t)bb*NN + n0)*NN;
  const _Float16* Bb = hT  + (size_t)bb*DD*NN;
  v8f acc[6];
#pragma unroll
  for (int f=0; f<6; f++)
#pragma unroll
    for (int j=0;j<8;j++) acc[f][j] = 0.f;
#pragma unroll 4
  for (int kc=0; kc<64; kc++) {
    v16h a = load_afrag_f32(Ab + kc*32, NN, lo, hi);
#pragma unroll
    for (int f=0; f<6; f++)
      acc[f] = wmma_f16(a, load_bfrag_f16(Bb + (size_t)(f*16)*NN + kc*32, NN, lo, hi), acc[f]);
  }
#pragma unroll
  for (int f=0; f<6; f++)
    store_cfrag_f32(aOut + (size_t)row0*DD + f*16, DD, acc[f], lo, hi);
}

// ---------------- GRU part 1: z, r*h (f16), a@Wh0+bh ----------------
__global__ __launch_bounds__(BLOCKT) void k_gru_zr(
    const float* __restrict__ aB, const float* __restrict__ h,
    const _Float16* __restrict__ Wg,   // [6][96][96] (transposed blocks z0,z1,r0,r1,h0,h1)
    const float* __restrict__ bg,      // [6][96]
    float* __restrict__ zO, _Float16* __restrict__ rhO, float* __restrict__ ahO)
{
  const int wave = threadIdx.x >> 5, lane = threadIdx.x & 31;
  const int lo = lane & 15, hi = lane >> 4;
  const int row0 = (blockIdx.x*WAVES + wave)*16;
  v16h aF[3], hF[3];
#pragma unroll
  for (int kc=0; kc<3; kc++) {
    aF[kc] = load_afrag_f32(aB + (size_t)row0*DD + kc*32, DD, lo, hi);
    hF[kc] = load_afrag_f32(h  + (size_t)row0*DD + kc*32, DD, lo, hi);
  }
  // z = sigmoid(a@Wz0 + h@Wz1 + bz0 + bz1)
  v8f z[6];
#pragma unroll
  for (int f=0; f<6; f++) { float bv = bg[0*DD+f*16+lo] + bg[1*DD+f*16+lo];
#pragma unroll
    for (int j=0;j<8;j++) z[f][j] = bv; }
#pragma unroll
  for (int kc=0; kc<3; kc++)
#pragma unroll
    for (int f=0; f<6; f++) {
      z[f] = wmma_f16(aF[kc], load_bfrag_f16(Wg + 0*DD*DD + (size_t)(f*16)*DD + kc*32, DD, lo, hi), z[f]);
      z[f] = wmma_f16(hF[kc], load_bfrag_f16(Wg + 1*DD*DD + (size_t)(f*16)*DD + kc*32, DD, lo, hi), z[f]);
    }
#pragma unroll
  for (int f=0; f<6; f++) {
#pragma unroll
    for (int j=0;j<8;j++) z[f][j] = sigmoidf_(z[f][j]);
    store_cfrag_f32(zO + (size_t)row0*DD + f*16, DD, z[f], lo, hi);
  }
  // r = sigmoid(a@Wr0 + h@Wr1 + br0 + br1); rh = r*h stored f16 row-major
#pragma unroll
  for (int f=0; f<6; f++) {
    v8f r; float bv = bg[2*DD+f*16+lo] + bg[3*DD+f*16+lo];
#pragma unroll
    for (int j=0;j<8;j++) r[j] = bv;
#pragma unroll
    for (int kc=0; kc<3; kc++) {
      r = wmma_f16(aF[kc], load_bfrag_f16(Wg + 2*DD*DD + (size_t)(f*16)*DD + kc*32, DD, lo, hi), r);
      r = wmma_f16(hF[kc], load_bfrag_f16(Wg + 3*DD*DD + (size_t)(f*16)*DD + kc*32, DD, lo, hi), r);
    }
    v8f hc = load_cfrag_f32(h + (size_t)row0*DD + f*16, DD, lo, hi);
    _Float16* rp = rhO + (size_t)(row0 + 8*hi)*DD + f*16 + lo;
#pragma unroll
    for (int j=0;j<8;j++)
      rp[(size_t)j*DD] = (_Float16)(sigmoidf_(r[j]) * hc[j]);
  }
  // ah = a@Wh0 + bh0 + bh1
#pragma unroll
  for (int f=0; f<6; f++) {
    v8f ah; float bv = bg[4*DD+f*16+lo] + bg[5*DD+f*16+lo];
#pragma unroll
    for (int j=0;j<8;j++) ah[j] = bv;
#pragma unroll
    for (int kc=0; kc<3; kc++)
      ah = wmma_f16(aF[kc], load_bfrag_f16(Wg + 4*DD*DD + (size_t)(f*16)*DD + kc*32, DD, lo, hi), ah);
    store_cfrag_f32(ahO + (size_t)row0*DD + f*16, DD, ah, lo, hi);
  }
}

// ---------------- GRU part 2: hh = relu(mask*(ah + rh@Wh1)); h = hh*z + h*(1-z) ----------------
__global__ __launch_bounds__(BLOCKT) void k_gru_h(
    const float* __restrict__ ahB, const _Float16* __restrict__ rhB,
    const _Float16* __restrict__ Wh1T, const float* __restrict__ zB,
    const float* __restrict__ mask,
    float* __restrict__ h, _Float16* __restrict__ hT)
{
  const int wave = threadIdx.x >> 5, lane = threadIdx.x & 31;
  const int lo = lane & 15, hi = lane >> 4;
  const int row0 = (blockIdx.x*WAVES + wave)*16;
  v16h rF[3];
#pragma unroll
  for (int kc=0; kc<3; kc++)
    rF[kc] = load_afrag_f16(rhB + (size_t)row0*DD + kc*32, DD, lo, hi);
#pragma unroll
  for (int f=0; f<6; f++) {
    v8f acc = load_cfrag_f32(ahB + (size_t)row0*DD + f*16, DD, lo, hi);
#pragma unroll
    for (int kc=0; kc<3; kc++)
      acc = wmma_f16(rF[kc], load_bfrag_f16(Wh1T + (size_t)(f*16)*DD + kc*32, DD, lo, hi), acc);
    v8f zc = load_cfrag_f32(zB + (size_t)row0*DD + f*16, DD, lo, hi);
    v8f hc = load_cfrag_f32(h  + (size_t)row0*DD + f*16, DD, lo, hi);
#pragma unroll
    for (int j=0;j<8;j++) {
      int m = row0 + 8*hi + j;
      float mv = mask[m];
      float hhv = mv * acc[j]; hhv = hhv > 0.f ? hhv : 0.f;
      float hn = hhv * zc[j] + hc[j] * (1.f - zc[j]);
      h[(size_t)m*DD + f*16 + lo] = hn;
      hT[((size_t)(m>>11)*DD + f*16 + lo)*NN + (m & (NN-1))] = (_Float16)hn;
    }
  }
}

// ---------------- generic 96x96 GEMM + bias: out = A @ WT^T + bias ----------------
__global__ __launch_bounds__(BLOCKT) void k_gemm96(
    const float* __restrict__ A, const _Float16* __restrict__ WT,
    const float* __restrict__ bias, float* __restrict__ out)
{
  const int wave = threadIdx.x >> 5, lane = threadIdx.x & 31;
  const int lo = lane & 15, hi = lane >> 4;
  const int row0 = (blockIdx.x*WAVES + wave)*16;
  v16h aF[3];
#pragma unroll
  for (int kc=0; kc<3; kc++)
    aF[kc] = load_afrag_f32(A + (size_t)row0*DD + kc*32, DD, lo, hi);
#pragma unroll
  for (int f=0; f<6; f++) {
    v8f acc; float bv = bias[f*16+lo];
#pragma unroll
    for (int j=0;j<8;j++) acc[j] = bv;
#pragma unroll
    for (int kc=0; kc<3; kc++)
      acc = wmma_f16(aF[kc], load_bfrag_f16(WT + (size_t)(f*16)*DD + kc*32, DD, lo, hi), acc);
    store_cfrag_f32(out + (size_t)row0*DD + f*16, DD, acc, lo, hi);
  }
}

// ---------------- leaky-relu pair fusion -> f16 A-matrices ----------------
__global__ __launch_bounds__(256) void k_lrelu(
    const float* __restrict__ t0, const float* __restrict__ t1, const float* __restrict__ t2,
    _Float16* __restrict__ a0, _Float16* __restrict__ a1, _Float16* __restrict__ a2, int n)
{
  int i = blockIdx.x*256 + threadIdx.x;
  if (i < n) {
    float v0 = t0[i], v1 = t1[i], v2 = t2[i];
    float s0 = v0 + v1, s1 = v1 + v2, s2 = v2 + v0;
    a0[i] = (_Float16)(s0 > 0.f ? s0 : 0.2f*s0);
    a1[i] = (_Float16)(s1 > 0.f ? s1 : 0.2f*s1);
    a2[i] = (_Float16)(s2 > 0.f ? s2 : 0.2f*s2);
  }
}

// ---------------- attention fusion: out = sum_i a_i @ Watt_i + sum_i b_att_i ----------------
__global__ __launch_bounds__(BLOCKT) void k_att(
    const _Float16* __restrict__ a0, const _Float16* __restrict__ a1, const _Float16* __restrict__ a2,
    const _Float16* __restrict__ WattT, const float* __restrict__ batt,
    float* __restrict__ out)
{
  const int wave = threadIdx.x >> 5, lane = threadIdx.x & 31;
  const int lo = lane & 15, hi = lane >> 4;
  const int row0 = (blockIdx.x*WAVES + wave)*16;
  v8f acc[6];
#pragma unroll
  for (int f=0; f<6; f++) {
    float bv = batt[f*16+lo] + batt[DD+f*16+lo] + batt[2*DD+f*16+lo];
#pragma unroll
    for (int j=0;j<8;j++) acc[f][j] = bv;
  }
  const _Float16* As[3] = {a0, a1, a2};
#pragma unroll
  for (int i=0;i<3;i++)
#pragma unroll
    for (int kc=0; kc<3; kc++) {
      v16h aF = load_afrag_f16(As[i] + (size_t)row0*DD + kc*32, DD, lo, hi);
#pragma unroll
      for (int f=0; f<6; f++)
        acc[f] = wmma_f16(aF, load_bfrag_f16(WattT + (size_t)i*DD*DD + (size_t)(f*16)*DD + kc*32, DD, lo, hi), acc[f]);
    }
#pragma unroll
  for (int f=0; f<6; f++)
    store_cfrag_f32(out + (size_t)row0*DD + f*16, DD, acc[f], lo, hi);
}

// ---------------- host launcher ----------------
extern "C" void kernel_launch(void* const* d_in, const int* in_sizes, int n_in,
                              void* d_out, int out_size, void* d_ws, size_t ws_size,
                              hipStream_t stream)
{
  (void)in_sizes; (void)n_in; (void)out_size; (void)ws_size;
  const float* x        = (const float*)d_in[0];
  const float* adjs[3]  = {(const float*)d_in[1], (const float*)d_in[2], (const float*)d_in[3]};
  const float* masks[3] = {(const float*)d_in[4], (const float*)d_in[5], (const float*)d_in[6]};
  const float* W_enc = (const float*)d_in[7];
  const float* b_enc = (const float*)d_in[8];
  const float* W_ii  = (const float*)d_in[9];
  const float* b_ii  = (const float*)d_in[10];
  const float* W_gru = (const float*)d_in[11];
  const float* b_gru = (const float*)d_in[12];
  const float* W_att = (const float*)d_in[13];
  const float* b_att = (const float*)d_in[14];

  char* w = (char*)d_ws;
  size_t o = 0;
  auto alloc = [&](size_t bytes)->char* { char* p = w + o; o += (bytes + 255) & ~(size_t)255; return p; };
  _Float16* WencT = (_Float16*)alloc((size_t)3*DD*KP*2);
  _Float16* WgruT = (_Float16*)alloc((size_t)18*DD*DD*2);
  _Float16* WiiT  = (_Float16*)alloc((size_t)3*DD*DD*2);
  _Float16* WattT = (_Float16*)alloc((size_t)3*DD*DD*2);
  float* hBuf[3]; _Float16* hTBuf[3];
  for (int i=0;i<3;i++) hBuf[i]  = (float*)alloc((size_t)ROWS*DD*4);
  for (int i=0;i<3;i++) hTBuf[i] = (_Float16*)alloc((size_t)ROWS*DD*2);
  float* aBuf  = (float*)alloc((size_t)ROWS*DD*4);
  float* zBuf  = (float*)alloc((size_t)ROWS*DD*4);
  float* ahBuf = (float*)alloc((size_t)ROWS*DD*4);
  _Float16* rhBuf = (_Float16*)alloc((size_t)ROWS*DD*2);
  _Float16* aAtt[3];
  for (int i=0;i<3;i++) aAtt[i] = (_Float16*)alloc((size_t)ROWS*DD*2);

  const int prepTot = 3*DD*KP + 18*DD*DD + 3*DD*DD + 3*DD*DD;
  k_prep<<<(prepTot+255)/256, dim3(256), 0, stream>>>(W_enc, W_ii, W_gru, W_att,
                                                      WencT, WiiT, WgruT, WattT);
  for (int i=0;i<3;i++) {
    k_encode<<<GGRID, dim3(BLOCKT), 0, stream>>>(x, masks[i], WencT + (size_t)i*DD*KP,
                                                 b_enc + i*DD, hBuf[i], hTBuf[i]);
    for (int s=0;s<2;s++) {
      k_mp<<<GGRID, dim3(BLOCKT), 0, stream>>>(adjs[i], hTBuf[i], aBuf);
      k_gru_zr<<<GGRID, dim3(BLOCKT), 0, stream>>>(aBuf, hBuf[i],
                                                   WgruT + (size_t)i*6*DD*DD,
                                                   b_gru + (size_t)i*6*DD,
                                                   zBuf, rhBuf, ahBuf);
      k_gru_h<<<GGRID, dim3(BLOCKT), 0, stream>>>(ahBuf, rhBuf,
                                                  WgruT + (size_t)(i*6+5)*DD*DD,
                                                  zBuf, masks[i], hBuf[i], hTBuf[i]);
    }
  }
  float* tB[3] = {aBuf, zBuf, ahBuf};   // reuse GRU scratch for t0,t1,t2
  for (int i=0;i<3;i++)
    k_gemm96<<<GGRID, dim3(BLOCKT), 0, stream>>>(hBuf[i], WiiT + (size_t)i*DD*DD,
                                                 b_ii + i*DD, tB[i]);
  const int nTot = ROWS*DD;
  k_lrelu<<<(nTot+255)/256, dim3(256), 0, stream>>>(tB[0], tB[1], tB[2],
                                                    aAtt[0], aAtt[1], aAtt[2], nTot);
  k_att<<<GGRID, dim3(BLOCKT), 0, stream>>>(aAtt[0], aAtt[1], aAtt[2],
                                            WattT, b_att, (float*)d_out);
}
